// MoEFFN_67199058313909
// MI455X (gfx1250) — compile-verified, matrix operand order checked by reading, not run on hardware
//
#include <hip/hip_runtime.h>
#include <hip/hip_bf16.h>
#include <math.h>

// ---------------- problem constants ----------------
#define TTOK 16384
#define DIMD 512
#define HIDN 2048
#define NEXP 8
#define CAPN 5120          // int(1.25 * T * K / E)
#define MAXN (2 * CAPN)    // max tokens per expert list (K slots x cap)
#define NBIN 1024
#define TM   32            // tokens per FFN block
#define HC   64            // HID chunk width

typedef __bf16 bf16_t;
typedef __attribute__((ext_vector_type(16))) __bf16 v16bf;
typedef __attribute__((ext_vector_type(8)))  __bf16 v8bf;
typedef __attribute__((ext_vector_type(8)))  float  flt8;

// ---------------- ws layout ----------------
// int-indexed control region:
#define OFF_HIST    0        // int [2][8][1024] = 16384
#define OFF_THRESH  16384    // int [16]
#define OFF_QUOTA   16400    // int [16]
#define OFF_QCTR    16416    // int [16]
#define OFF_CNT     16432    // int [8]
#define OFF_LOAD    16440    // float [8]
#define OFF_IMP     16448    // float [8]
#define CTRL_INTS   16456
#define OFF_TOPIDX  16512    // int [T][2]
#define OFF_TOPW    49280    // float [T][2]
#define OFF_LTOK    82048    // int [8][MAXN]
#define OFF_LW      163968   // float [8][MAXN]
// byte-indexed bf16 region:
#define BF16_BASE   1048576                 // 1 MB
#define NELEM_X     ((size_t)TTOK * DIMD)            // 8388608
#define NELEM_W     ((size_t)NEXP * HIDN * DIMD)     // 8388608
#define WS_NEED     (BF16_BASE + 2 * (NELEM_X + 2 * NELEM_W))  // ~51.4 MB

// ---------------- helpers ----------------
__global__ __launch_bounds__(256) void zero_f_kernel(float* p, int n) {
    int i = blockIdx.x * 256 + threadIdx.x;
    if (i < n) p[i] = 0.0f;
}
__global__ __launch_bounds__(256) void zero_i_kernel(int* p, int n) {
    int i = blockIdx.x * 256 + threadIdx.x;
    if (i < n) p[i] = 0;
}
__global__ __launch_bounds__(256) void cvt_bf16_kernel(const float* __restrict__ src,
                                                       bf16_t* __restrict__ dst, int n8) {
    int i = blockIdx.x * 256 + threadIdx.x;
    if (i < n8) {
        flt8 v = *(const flt8*)(src + (size_t)i * 8);
        v8bf o;
        #pragma unroll
        for (int k = 0; k < 8; ++k) o[k] = (bf16_t)v[k];
        *(v8bf*)(dst + (size_t)i * 8) = o;
    }
}

// ---------------- router: logits -> softmax -> top2 -> hist ----------------
__global__ __launch_bounds__(256) void router_kernel(
    const float* __restrict__ x, const float* __restrict__ Wr,
    int* __restrict__ top_idx, float* __restrict__ top_w,
    int* __restrict__ hist, float* __restrict__ imp)
{
    __shared__ float simp[NEXP];
    int tid = threadIdx.x;
    int t = blockIdx.x * 256 + tid;
    if (tid < NEXP) simp[tid] = 0.0f;
    __syncthreads();
    if (t < TTOK) {
        float acc[NEXP];
        #pragma unroll
        for (int e = 0; e < NEXP; ++e) acc[e] = 0.0f;
        const float* xr = x + (size_t)t * DIMD;
        for (int d = 0; d < DIMD; d += 4) {
            float4 xv = *(const float4*)(xr + d);
            #pragma unroll
            for (int e = 0; e < NEXP; ++e) {
                float4 wv = *(const float4*)(Wr + e * DIMD + d);
                acc[e] += xv.x * wv.x + xv.y * wv.y + xv.z * wv.z + xv.w * wv.w;
            }
        }
        float m = acc[0];
        #pragma unroll
        for (int e = 1; e < NEXP; ++e) m = fmaxf(m, acc[e]);
        float p[NEXP], s = 0.0f;
        #pragma unroll
        for (int e = 0; e < NEXP; ++e) { p[e] = __expf(acc[e] - m); s += p[e]; }
        float inv = 1.0f / s;
        #pragma unroll
        for (int e = 0; e < NEXP; ++e) p[e] *= inv;
        #pragma unroll
        for (int e = 0; e < NEXP; ++e) atomicAdd(&simp[e], p[e]);
        int i0 = 0; float w0 = p[0];
        #pragma unroll
        for (int e = 1; e < NEXP; ++e) if (p[e] > w0) { w0 = p[e]; i0 = e; }
        int i1 = -1; float w1 = -1.0f;
        #pragma unroll
        for (int e = 0; e < NEXP; ++e) if (e != i0 && p[e] > w1) { w1 = p[e]; i1 = e; }
        float s2 = w0 + w1 + 1e-8f;
        float nw0 = w0 / s2, nw1 = w1 / s2;
        top_idx[t * 2 + 0] = i0; top_idx[t * 2 + 1] = i1;
        top_w[t * 2 + 0] = nw0;  top_w[t * 2 + 1] = nw1;
        int b0 = min(NBIN - 1, (int)(nw0 * (float)NBIN));
        int b1 = min(NBIN - 1, (int)(nw1 * (float)NBIN));
        atomicAdd(&hist[(0 * NEXP + i0) * NBIN + b0], 1);
        atomicAdd(&hist[(1 * NEXP + i1) * NBIN + b1], 1);
    }
    __syncthreads();
    if (tid < NEXP) atomicAdd(&imp[tid], simp[tid]);
}

// ---------------- capacity threshold per (slot, expert) ----------------
__global__ void thresh_kernel(const int* __restrict__ hist,
                              int* __restrict__ thresh, int* __restrict__ quota)
{
    int id = threadIdx.x;
    if (id >= 16) return;
    int cum = 0, th = -1, q = 0;
    const int* h = hist + id * NBIN;
    for (int b = NBIN - 1; b >= 0; --b) {
        int c = h[b];
        if (cum + c >= CAPN) { th = b; q = CAPN - cum; break; }
        cum += c;
    }
    thresh[id] = th; quota[id] = q;
}

// ---------------- keep/drop, renorm, build expert lists ----------------
__global__ __launch_bounds__(256) void assign_kernel(
    const int* __restrict__ top_idx, const float* __restrict__ top_w,
    const int* __restrict__ thresh, const int* __restrict__ quota,
    int* __restrict__ qctr, int* __restrict__ cnt,
    int* __restrict__ ltok, float* __restrict__ lwt, float* __restrict__ load_sum)
{
    int t = blockIdx.x * 256 + threadIdx.x;
    if (t >= TTOK) return;
    int   ke[2]; float kw[2]; bool kp[2];
    #pragma unroll
    for (int k = 0; k < 2; ++k) {
        float w = top_w[t * 2 + k];
        int   e = top_idx[t * 2 + k];
        int   b = min(NBIN - 1, (int)(w * (float)NBIN));
        int  th = thresh[k * NEXP + e];
        bool keep;
        if (b > th) keep = true;
        else if (b == th) {
            int pos = atomicAdd(&qctr[k * NEXP + e], 1);
            keep = pos < quota[k * NEXP + e];
        } else keep = false;
        ke[k] = e; kw[k] = w; kp[k] = keep;
    }
    float s = (kp[0] ? kw[0] : 0.0f) + (kp[1] ? kw[1] : 0.0f);
    float denom = fmaxf(s, 1e-8f);
    #pragma unroll
    for (int k = 0; k < 2; ++k) {
        if (kp[k]) {
            float wn = kw[k] / denom;
            int pos = atomicAdd(&cnt[ke[k]], 1);
            if (pos < MAXN) {
                ltok[ke[k] * MAXN + pos] = t;
                lwt[ke[k] * MAXN + pos] = wn;
            }
            atomicAdd(&load_sum[ke[k]], wn);
        }
    }
}

// ---------------- WMMA fragment loaders ----------------
// 16-bit A/B layout (ISA 7.12.2): lane half selects K+{0..7,16..23} / K+{8..15,24..31}
__device__ __forceinline__ v16bf fragLDS(const bf16_t* base, int stride, int row, int kb) {
    const bf16_t* p = base + row * stride + kb;
    v8bf lo = *(const v8bf*)p;
    v8bf hi = *(const v8bf*)(p + 16);
    return __builtin_shufflevector(lo, hi, 0,1,2,3,4,5,6,7,8,9,10,11,12,13,14,15);
}
__device__ __forceinline__ v16bf fragGb16(const bf16_t* __restrict__ p, int kb) {
    v8bf lo = *(const v8bf*)(p + kb);
    v8bf hi = *(const v8bf*)(p + kb + 16);
    return __builtin_shufflevector(lo, hi, 0,1,2,3,4,5,6,7,8,9,10,11,12,13,14,15);
}
__device__ __forceinline__ v16bf fragGf32(const float* __restrict__ p, int kb) {
    flt8 a0 = *(const flt8*)(p + kb);
    flt8 a1 = *(const flt8*)(p + kb + 16);
    v16bf r;
    #pragma unroll
    for (int i = 0; i < 8; ++i) { r[i] = (bf16_t)a0[i]; r[i + 8] = (bf16_t)a1[i]; }
    return r;
}

// ---------------- fused expert FFN: fc1 -> gelu -> fc2 -> weighted scatter ----------------
// Block: 256 threads = 8 waves = 2 m-tiles x 4 col-groups; TM=32 tokens; HID chunk 64.
template <bool FAST>
__global__ __launch_bounds__(256) void moe_ffn_kernel(
    const float* __restrict__ x,  const bf16_t* __restrict__ xb,
    const float* __restrict__ W1, const bf16_t* __restrict__ W1b,
    const float* __restrict__ b1,
    const float* __restrict__ W2, const bf16_t* __restrict__ W2b,
    const float* __restrict__ b2,
    const int* __restrict__ cnt, const int* __restrict__ ltok,
    const float* __restrict__ lwt, float* __restrict__ out)
{
    const int e  = blockIdx.y;
    const int m0 = blockIdx.x * TM;
    const int nc = cnt[e];
    if (m0 >= nc) return;

    __shared__ __align__(16) bf16_t xs[TM][520];     // 32 tokens x 512 bf16 (padded)
    __shared__ __align__(16) bf16_t hs[TM][HC + 8];  // 32 tokens x 64 hid chunk
    __shared__ int   stok[TM];
    __shared__ float sw[TM];

    const int tid  = threadIdx.x;
    const int wave = tid >> 5;
    const int lane = tid & 31;
    const int lh   = lane >> 4;
    const int lr   = lane & 15;
    const int kboff = lh * 8;
    const int mt   = wave & 1;      // m-tile (rows mt*16 .. mt*16+15)
    const int wc   = wave >> 1;     // col group 0..3

    if (tid < TM) {
        int gm = m0 + tid;
        if (gm < nc) { stok[tid] = ltok[e * MAXN + gm]; sw[tid] = lwt[e * MAXN + gm]; }
        else         { stok[tid] = ltok[e * MAXN];      sw[tid] = 0.0f; }
    }
    __syncthreads();
    // stage x tile -> bf16 LDS
    if constexpr (FAST) {
        for (int idx = tid * 8; idx < TM * DIMD; idx += 256 * 8) {
            int r = idx >> 9, c = idx & (DIMD - 1);
            *(v8bf*)&xs[r][c] = *(const v8bf*)(xb + (size_t)stok[r] * DIMD + c);
        }
    } else {
        for (int idx = tid; idx < TM * DIMD; idx += 256) {
            int r = idx >> 9, c = idx & (DIMD - 1);
            xs[r][c] = (bf16_t)x[(size_t)stok[r] * DIMD + c];
        }
    }
    __builtin_prefetch(W1 + (size_t)e * HIDN * DIMD, 0, 3);   // global_prefetch path
    __syncthreads();

    flt8 y[8];
    #pragma unroll
    for (int j = 0; j < 8; ++j) y[j] = (flt8){0.f,0.f,0.f,0.f,0.f,0.f,0.f,0.f};

    for (int ch = 0; ch < HIDN / HC; ++ch) {
        // ---- fc1: each wave computes one 16x16 h tile over K=512 ----
        const int n0 = ch * HC + wc * 16;     // hid col tile base
        const bf16_t* w1rb = nullptr; const float* w1rf = nullptr;
        if constexpr (FAST) w1rb = W1b + ((size_t)e * HIDN + (n0 + lr)) * DIMD;
        else                w1rf = W1  + ((size_t)e * HIDN + (n0 + lr)) * DIMD;
        flt8 c1 = (flt8){0.f,0.f,0.f,0.f,0.f,0.f,0.f,0.f};
        #pragma unroll
        for (int kk = 0; kk < DIMD; kk += 32) {
            v16bf a = fragLDS(&xs[0][0], 520, mt * 16 + lr, kk + kboff);
            v16bf b;
            if constexpr (FAST) b = fragGb16(w1rb, kk + kboff);
            else                b = fragGf32(w1rf, kk + kboff);
            c1 = __builtin_amdgcn_wmma_f32_16x16x32_bf16(false, a, false, b,
                                                         (short)0, c1, false, false);
        }
        const float bb1 = b1[(size_t)e * HIDN + n0 + lr];
        #pragma unroll
        for (int i = 0; i < 8; ++i) {
            float v = c1[i] + bb1;
            float g = 0.5f * v * (1.0f + erff(v * 0.70710678118654752f));
            hs[mt * 16 + lh * 8 + i][wc * 16 + lr] = (bf16_t)g;
        }
        __syncthreads();
        // ---- fc2: each wave accumulates 8 y tiles over this 64-wide hid chunk ----
        #pragma unroll
        for (int j = 0; j < 8; ++j) {
            const int n = wc * 128 + j * 16 + lr;     // out col
            const bf16_t* w2rb = nullptr; const float* w2rf = nullptr;
            if constexpr (FAST) w2rb = W2b + ((size_t)e * DIMD + n) * HIDN + ch * HC;
            else                w2rf = W2  + ((size_t)e * DIMD + n) * HIDN + ch * HC;
            #pragma unroll
            for (int kk = 0; kk < HC; kk += 32) {
                v16bf a = fragLDS(&hs[0][0], HC + 8, mt * 16 + lr, kk + kboff);
                v16bf b;
                if constexpr (FAST) b = fragGb16(w2rb, kk + kboff);
                else                b = fragGf32(w2rf, kk + kboff);
                y[j] = __builtin_amdgcn_wmma_f32_16x16x32_bf16(false, a, false, b,
                                                               (short)0, y[j], false, false);
            }
        }
        __syncthreads();
    }

    // ---- epilogue: weighted atomic scatter into out ----
    #pragma unroll
    for (int j = 0; j < 8; ++j) {
        int n = wc * 128 + j * 16 + lr;
        float bb2 = b2[(size_t)e * DIMD + n];
        #pragma unroll
        for (int i = 0; i < 8; ++i) {
            int m = mt * 16 + lh * 8 + i;
            if (m0 + m < nc) {
                atomicAdd(&out[(size_t)stok[m] * DIMD + n], sw[m] * (y[j][i] + bb2));
            }
        }
    }
}

// ---------------- aux loss ----------------
__global__ void aux_kernel(const float* __restrict__ imp,
                           const float* __restrict__ load_sum, float* __restrict__ out)
{
    if (threadIdx.x == 0) {
        float a = 0.0f;
        #pragma unroll
        for (int e = 0; e < NEXP; ++e)
            a += (imp[e] / (float)TTOK) * (load_sum[e] / (float)TTOK) * (float)NEXP;
        out[(size_t)TTOK * DIMD] = a;
    }
}

// ---------------- host launch ----------------
extern "C" void kernel_launch(void* const* d_in, const int* in_sizes, int n_in,
                              void* d_out, int out_size, void* d_ws, size_t ws_size,
                              hipStream_t stream) {
    const float* x  = (const float*)d_in[0];
    const float* Wr = (const float*)d_in[1];
    const float* W1 = (const float*)d_in[2];
    const float* b1 = (const float*)d_in[3];
    const float* W2 = (const float*)d_in[4];
    const float* b2 = (const float*)d_in[5];
    float* out = (float*)d_out;

    int* wsI = (int*)d_ws;
    int*   hist     = wsI + OFF_HIST;
    int*   thresh   = wsI + OFF_THRESH;
    int*   quota    = wsI + OFF_QUOTA;
    int*   qctr     = wsI + OFF_QCTR;
    int*   cnt      = wsI + OFF_CNT;
    float* load_sum = (float*)(wsI + OFF_LOAD);
    float* imp      = (float*)(wsI + OFF_IMP);
    int*   top_idx  = wsI + OFF_TOPIDX;
    float* top_w    = (float*)(wsI + OFF_TOPW);
    int*   ltok     = wsI + OFF_LTOK;
    float* lwt      = (float*)(wsI + OFF_LW);

    bf16_t* xb  = (bf16_t*)((char*)d_ws + BF16_BASE);
    bf16_t* w1b = xb + NELEM_X;
    bf16_t* w2b = w1b + NELEM_W;
    const bool fast = ws_size >= WS_NEED;   // fixed per process -> deterministic

    // zero output and control block (fresh every call; graph-replay safe)
    zero_f_kernel<<<(out_size + 255) / 256, 256, 0, stream>>>(out, out_size);
    zero_i_kernel<<<(CTRL_INTS + 255) / 256, 256, 0, stream>>>(wsI, CTRL_INTS);

    if (fast) {   // pre-convert x, W1, W2 to bf16 (8 elems/thread)
        cvt_bf16_kernel<<<(int)(NELEM_X / 8 / 256), 256, 0, stream>>>(x,  xb,  (int)(NELEM_X / 8));
        cvt_bf16_kernel<<<(int)(NELEM_W / 8 / 256), 256, 0, stream>>>(W1, w1b, (int)(NELEM_W / 8));
        cvt_bf16_kernel<<<(int)(NELEM_W / 8 / 256), 256, 0, stream>>>(W2, w2b, (int)(NELEM_W / 8));
    }

    router_kernel<<<TTOK / 256, 256, 0, stream>>>(x, Wr, top_idx, top_w, hist, imp);
    thresh_kernel<<<1, 16, 0, stream>>>(hist, thresh, quota);
    assign_kernel<<<TTOK / 256, 256, 0, stream>>>(top_idx, top_w, thresh, quota,
                                                  qctr, cnt, ltok, lwt, load_sum);

    dim3 ffn_grid(MAXN / TM, NEXP);
    if (fast) {
        moe_ffn_kernel<true><<<ffn_grid, 256, 0, stream>>>(
            x, xb, W1, w1b, b1, W2, w2b, b2, cnt, ltok, lwt, out);
    } else {
        moe_ffn_kernel<false><<<ffn_grid, 256, 0, stream>>>(
            x, xb, W1, w1b, b1, W2, w2b, b2, cnt, ltok, lwt, out);
    }

    aux_kernel<<<1, 32, 0, stream>>>(imp, load_sum, out);
}
